// Feat_Similarity_18107582120403
// MI455X (gfx1250) — matrix operand order repaired and checked by reading, not compiled, not run on hardware
//
#include <hip/hip_runtime.h>
#include <hip/hip_bf16.h>

typedef _Float16 h4   __attribute__((ext_vector_type(4)));
typedef _Float16 v8h  __attribute__((ext_vector_type(8)));
typedef _Float16 v16h __attribute__((ext_vector_type(16)));
typedef float    v8f  __attribute__((ext_vector_type(8)));

#define B_ 128
#define N_ 2048
#define D_ 128
#define GSZ (B_ * B_)         // 16384 elements of G
#define LSTRIDE 72            // 64 f16 + 8 pad -> 144B rows, bank-conflict free

__device__ __forceinline__ v16h cat8(v8h lo, v8h hi) {
  return __builtin_shufflevector(lo, hi, 0,1,2,3,4,5,6,7,8,9,10,11,12,13,14,15);
}

__global__ void feat_sim_zero(float* __restrict__ p, int n) {
  int i = blockIdx.x * 256 + threadIdx.x;
  if (i < n) p[i] = 0.f;
}

// Per-WG partial of G[i,j] = sum_{n,d} x[i,n,d] * softmax_d(x[j,n,:])[d]
// (K-split over n across gridDim.x WGs, private partial buffers, no atomics)
// C[i] = sum_n logsumexp_d(x[i,n,:])  (tiny, atomics)
__global__ __launch_bounds__(256) void feat_sim_gemm(const float* __restrict__ x,
                                                     float* __restrict__ P,
                                                     float* __restrict__ C) {
  __shared__ _Float16 XB[B_ * LSTRIDE];   // f16 x tile   [128][64+pad]
  __shared__ _Float16 QB[B_ * LSTRIDE];   // f16 q tile   [128][64+pad]

  const int tid   = threadIdx.x;
  const int wave  = tid >> 5;             // 0..7 -> i-tile
  const int lane  = tid & 31;
  const int half8 = lane >> 4;            // lane group 0..15 / 16..31
  const int m16   = lane & 15;
  const int row   = tid >> 1;             // sample 0..127
  const int dsel  = tid & 1;              // which 64-col half of d this thread owns
  const int chunk = N_ / gridDim.x;       // n-rows per WG

  v8f acc[8];
#pragma unroll
  for (int t = 0; t < 8; ++t) acc[t] = (v8f){0,0,0,0,0,0,0,0};
  float cAcc = 0.f;

  const size_t rowoff = (size_t)row * (size_t)(N_ * D_)
                      + (size_t)blockIdx.x * chunk * D_
                      + (size_t)dsel * 64;
  const float4* __restrict__ base = (const float4*)(x + rowoff);

  for (int nn = 0; nn < chunk; ++nn) {
    // ---- prefetch next iteration's 256B region (global_prefetch_b8) ----
    if (nn + 1 < chunk) {
      const char* np = (const char*)(base + (nn + 1) * (D_ / 4));
      __builtin_prefetch(np, 0, 3);
      __builtin_prefetch(np + 128, 0, 3);
    }
    // ---- load 64 floats of this row-half into registers ----
    float4 v[16];
#pragma unroll
    for (int c = 0; c < 16; ++c) v[c] = base[nn * (D_ / 4) + c];

    // ---- row max (combine with partner lane) ----
    float mx = v[0].x;
#pragma unroll
    for (int c = 0; c < 16; ++c)
      mx = fmaxf(mx, fmaxf(fmaxf(v[c].x, v[c].y), fmaxf(v[c].z, v[c].w)));
    mx = fmaxf(mx, __shfl_xor(mx, 1, 32));

    // ---- exp + sum; keep x as f16, overwrite v with exp ----
    h4 xh[16];
    float s = 0.f;
#pragma unroll
    for (int c = 0; c < 16; ++c) {
      xh[c] = (h4){(_Float16)v[c].x, (_Float16)v[c].y,
                   (_Float16)v[c].z, (_Float16)v[c].w};
      float4 e;
      e.x = __expf(v[c].x - mx); e.y = __expf(v[c].y - mx);
      e.z = __expf(v[c].z - mx); e.w = __expf(v[c].w - mx);
      s += (e.x + e.y) + (e.z + e.w);
      v[c] = e;
    }
    s += __shfl_xor(s, 1, 32);
    const float inv = __frcp_rn(s);
    if (dsel == 0) cAcc += mx + __logf(s);   // logsumexp accumulation

    // ---- two phases over the two 64-wide d-halves ----
#pragma unroll
    for (int ph = 0; ph < 2; ++ph) {
      __syncthreads();                       // previous WMMA reads done
      if (dsel == ph) {
        _Float16* xr = &XB[row * LSTRIDE];
        _Float16* qr = &QB[row * LSTRIDE];
#pragma unroll
        for (int c = 0; c < 16; ++c) {
          *(h4*)(xr + c * 4) = xh[c];
          h4 q = {(_Float16)(v[c].x * inv), (_Float16)(v[c].y * inv),
                  (_Float16)(v[c].z * inv), (_Float16)(v[c].w * inv)};
          *(h4*)(qr + c * 4) = q;
        }
      }
      __syncthreads();

      // ---- WMMA: wave owns i-tile = wave; 8 j-tiles x 2 k-steps of 32 ----
      // 16-bit A 16x32 layout: lanes 0-15 hold K 0..7 & 16..23; lanes 16-31 K 8..15 & 24..31
      const int arow = (wave * 16 + m16) * LSTRIDE;
      v16h af[2];
#pragma unroll
      for (int kt = 0; kt < 2; ++kt)
        af[kt] = cat8(*(const v8h*)(&XB[arow + kt * 32 + half8 * 8]),
                      *(const v8h*)(&XB[arow + kt * 32 + 16 + half8 * 8]));
#pragma unroll
      for (int kt = 0; kt < 2; ++kt) {
        // 16-bit B 32x16 layout: lane = column j; lanes 0-15 K 0..7 & 16..23, lanes 16-31 K 8..15 & 24..31
        // Issue all 16 ds_load_b128 into distinct live registers first, so the
        // scheduler can overlap DS latency with the WMMA stream (partial waits).
        v16h bf[8];
#pragma unroll
        for (int jt = 0; jt < 8; ++jt) {
          const int nb = (jt * 16 + m16) * LSTRIDE + kt * 32 + half8 * 8;
          ((v8h*)&bf[jt])[0] = *(const v8h*)(&QB[nb]);
          ((v8h*)&bf[jt])[1] = *(const v8h*)(&QB[nb + 16]);
        }
#pragma unroll
        for (int jt = 0; jt < 8; ++jt) {
          acc[jt] = __builtin_amdgcn_wmma_f32_16x16x32_f16(
              false, af[kt], false, bf[jt], (short)0, acc[jt], false, false);
        }
      }
    }
  }

  // ---- commit: private partial slice, plain stores (no atomics) ----
  if (dsel == 0) atomicAdd(&C[row], cAcc);
  float* __restrict__ Pw = P + (size_t)blockIdx.x * GSZ;
#pragma unroll
  for (int jt = 0; jt < 8; ++jt) {
#pragma unroll
    for (int r = 0; r < 8; ++r) {
      // 16x16 f32 C/D layout: VGPR r, lanes 0-15 -> M=r, lanes 16-31 -> M=r+8; N=lane%16
      int gi = wave * 16 + r + half8 * 8;
      int gj = jt * 16 + m16;
      Pw[gi * B_ + gj] = acc[jt][r];
    }
  }
}

// G[e] = sum_w P[w][e]; 16 blocks x 256 threads x float4 = 16384 elements
__global__ void feat_sim_reduce(const float* __restrict__ P, float* __restrict__ G,
                                int nwg) {
  const int gid = blockIdx.x * 256 + threadIdx.x;       // 0..4095
  const float4* p = (const float4*)P + gid;
  float4 s = {0.f, 0.f, 0.f, 0.f};
  for (int w = 0; w < nwg; ++w) {
    float4 t = p[(size_t)w * (GSZ / 4)];
    s.x += t.x; s.y += t.y; s.z += t.z; s.w += t.w;
  }
  ((float4*)G)[gid] = s;
}

__global__ void feat_sim_final(const float* __restrict__ G, const float* __restrict__ C,
                               const int* __restrict__ lab, float* __restrict__ out) {
  __shared__ float red[256];
  const int tid = threadIdx.x;
  const float scale = 1.0f / (float)(N_ * D_);
  float local = 0.f;
  for (int p = tid; p < B_ * B_; p += 256) {
    int i = p >> 7, j = p & (B_ - 1);
    if (i <= j) {
      // kl[i,j] = (ent[j] - cross[i,j]) / ND = (G[j,j]-C[j] - G[i,j]+C[i]) / ND
      float kl = (G[j * B_ + j] - C[j] - G[i * B_ + j] + C[i]) * scale;
      local += (lab[i] == lab[j]) ? kl : (1.0f - kl);
    }
  }
  red[tid] = local;
  __syncthreads();
  for (int s = 128; s > 0; s >>= 1) {
    if (tid < s) red[tid] += red[tid + s];
    __syncthreads();
  }
  if (tid == 0) out[0] = red[0];
}

extern "C" void kernel_launch(void* const* d_in, const int* in_sizes, int n_in,
                              void* d_out, int out_size, void* d_ws, size_t ws_size,
                              hipStream_t stream) {
  const float* x   = (const float*)d_in[0];   // [128, 2048, 128] f32
  const int*   lab = (const int*)d_in[1];     // [128] (jnp.int64 -> int32 without x64)

  // Largest power-of-two grid whose partial buffers fit the workspace
  // (ws_size is fixed across calls -> deterministic choice).
  int nwg = 256;
  while (nwg > 8 && ((size_t)nwg * GSZ + GSZ + B_) * sizeof(float) > ws_size)
    nwg >>= 1;

  float* P = (float*)d_ws;                    // nwg * 16384 partial G's
  float* G = P + (size_t)nwg * GSZ;           // 16384 reduced G
  float* C = G + GSZ;                         // 128

  feat_sim_zero<<<1, 128, 0, stream>>>(C, B_);
  feat_sim_gemm<<<nwg, 256, 0, stream>>>(x, P, C);
  feat_sim_reduce<<<GSZ / (256 * 4), 256, 0, stream>>>(P, G, nwg);
  feat_sim_final<<<1, 256, 0, stream>>>(G, C, lab, (float*)d_out);
}
